// MultiHeadAttention_28355374088946
// MI455X (gfx1250) — matrix operand order repaired and checked by reading, not compiled, not run on hardware
//
#include <hip/hip_runtime.h>

// ---------------------------------------------------------------------------
// MI455X (gfx1250) multi-head attention, bf16 WMMA pipeline.
//   0) transpose+convert the four weights to bf16 wT[N][K] (one-time)
//   1) gemm(mode=1): q = (Q @ w_q) * 0.125  -> bf16 [B,H,S,64]
//   2) gemm(mode=1): k = (K @ w_k)          -> bf16 [B,H,S,64]
//   3) gemm(mode=3): vT = (V @ w_v)^T       -> bf16 [B,H,64,S]
//   4) flash attention (online softmax), WMMA QK^T / PV, double-buffered
//      async-to-LDS K/V^T tiles -> x fp32 [B,S,1024]
//   5) gemm(mode=0): out = x @ w_o          -> fp32 d_out
// Matrix math: v_wmma_f32_16x16x32_bf16 (wave32). Tile staging through LDS,
// B-tiles / K / V^T via GLOBAL_LOAD_ASYNC_TO_LDS_B128 when available.
// ---------------------------------------------------------------------------

typedef __attribute__((ext_vector_type(16))) __bf16 v16bf;
typedef __attribute__((ext_vector_type(8)))  float  v8f;
typedef int v4i __attribute__((vector_size(4 * sizeof(int))));

#if __has_builtin(__builtin_amdgcn_global_load_async_to_lds_b128)
#define ASYNC_LDS 1
typedef v4i __attribute__((address_space(1)))* gvec_ptr;  // global (device)
typedef v4i __attribute__((address_space(3)))* lvec_ptr;  // LDS
#endif

// 16-byte global->LDS copy; async (ASYNCcnt-tracked) when the builtin exists.
__device__ __forceinline__ void copy16_g2l(const __bf16* g, __bf16* l) {
#ifdef ASYNC_LDS
  __builtin_amdgcn_global_load_async_to_lds_b128((gvec_ptr)(g), (lvec_ptr)(l), 0, 0);
#else
  *reinterpret_cast<uint4*>(l) = *reinterpret_cast<const uint4*>(g);
#endif
}

__device__ __forceinline__ void wait_async() {
#ifdef ASYNC_LDS
#if __has_builtin(__builtin_amdgcn_s_wait_asynccnt)
  __builtin_amdgcn_s_wait_asynccnt(0);
#else
  asm volatile("s_wait_asynccnt 0" ::: "memory");
#endif
#endif
}

union FragBF { uint4 u[2]; v16bf v; };

__device__ __forceinline__ v8f v8f_zero() {
  v8f z;
#pragma unroll
  for (int i = 0; i < 8; ++i) z[i] = 0.0f;
  return z;
}

// A-matrix fragment (16xK slice, 16-bit): lane holds row (lane&15),
// two 16B chunks at k = (lane>=16 ? 8 : 0) and +16.
__device__ __forceinline__ v16bf load_fragA(const __bf16* base, int stride, int lane) {
  const int row = lane & 15;
  const int kb  = (lane >> 4) << 3;  // 0 or 8
  FragBF f;
  f.u[0] = *reinterpret_cast<const uint4*>(base + row * stride + kb);
  f.u[1] = *reinterpret_cast<const uint4*>(base + row * stride + kb + 16);
  return f.v;
}

// B-matrix fragment (Kx16 slice, 16-bit): lane holds column (lane&15),
// two 16B chunks at k = (lane>=16 ? 8 : 0) and +16. LDS tile stored k-innermost.
__device__ __forceinline__ v16bf load_fragB(const __bf16* base, int stride, int lane) {
  const int n  = lane & 15;
  const int kb = (lane >> 4) << 3;  // 0 or 8
  FragBF f;
  const uint4* p = reinterpret_cast<const uint4*>(base + n * stride + kb);
  f.u[0] = p[0];
  f.u[1] = p[2];
  return f.v;
}

// ---------------------------------------------------------------------------
// One-time weight transpose + fp32->bf16: wt[n][k] = (bf16) w[k][n].
// 32x32 tiles through LDS so both global sides stay coalesced.
// ---------------------------------------------------------------------------
__global__ __launch_bounds__(256) void transpose_convert(
    const float* __restrict__ w, __bf16* __restrict__ wt, int K, int N) {
  __shared__ float tile[32][33];
  const int n0 = blockIdx.x * 32;
  const int k0 = blockIdx.y * 32;
  const int tx = threadIdx.x & 31;
  const int ty = threadIdx.x >> 5;  // 0..7
#pragma unroll
  for (int i = 0; i < 32; i += 8)
    tile[ty + i][tx] = w[(size_t)(k0 + ty + i) * N + n0 + tx];
  __syncthreads();
#pragma unroll
  for (int i = 0; i < 32; i += 8)
    wt[(size_t)(n0 + ty + i) * K + k0 + tx] = (__bf16)tile[tx][ty + i];
}

// ---------------------------------------------------------------------------
// GEMM: C = A[MxK](fp32) * Bt[NxK](bf16, pre-transposed)
//   mode 0: fp32 C[MxN]
//   mode 1: bf16 head-split  [(b*H+h)*S + s]*64 + d   (scaled; q/k)
//   mode 3: bf16 head-transposed [(b*H+h)*64 + d]*S + s  (v^T)
// Block tile 128x128, 8 waves (4x2), each wave 32x64 = 8 WMMA accum tiles.
// ---------------------------------------------------------------------------
namespace cfg {
constexpr int TM = 128, TN = 128, TK = 32, PAD = 8;
constexpr int Sq = 2048, Hh = 16;
}

__global__ __launch_bounds__(256) void gemm_bf16_wmma(
    const float* __restrict__ A, const __bf16* __restrict__ Bt,
    void* __restrict__ Cout, int M, int N, int K, int mode, float scale) {
  using namespace cfg;
  __shared__ __align__(16) __bf16 ldsA[TM][TK + PAD];  // [m][k]
  __shared__ __align__(16) __bf16 ldsB[TN][TK + PAD];  // [n][k]

  const int tid  = threadIdx.x;
  const int lane = tid & 31;
  const int wave = tid >> 5;
  const int wm   = wave >> 1;  // 0..3 -> 32-row strip
  const int wn   = wave & 1;   // 0..1 -> 64-col strip
  const int m0   = blockIdx.y * TM;
  const int n0   = blockIdx.x * TN;

  v8f acc[2][4];
#pragma unroll
  for (int i = 0; i < 2; ++i)
#pragma unroll
    for (int j = 0; j < 4; ++j) acc[i][j] = v8f_zero();

  const int arow = tid >> 1;        // 0..127
  const int acol = (tid & 1) << 4;  // 0 or 16
  const int bn   = tid >> 1;        // 0..127
  const int bks  = (tid & 1) << 4;  // 0 or 16

  for (int kt = 0; kt < K; kt += TK) {
    // ---- stage A tile: global fp32 -> cvt -> LDS bf16 (16 elems/thread)
    {
      const float4* g =
          reinterpret_cast<const float4*>(A + (size_t)(m0 + arow) * K + kt + acol);
      float4 f0 = g[0], f1 = g[1], f2 = g[2], f3 = g[3];
      const float fv[16] = {f0.x, f0.y, f0.z, f0.w, f1.x, f1.y, f1.z, f1.w,
                            f2.x, f2.y, f2.z, f2.w, f3.x, f3.y, f3.z, f3.w};
      union { __bf16 b[16]; uint4 u[2]; } pk;
#pragma unroll
      for (int i = 0; i < 16; ++i) pk.b[i] = (__bf16)fv[i];
      uint4* d = reinterpret_cast<uint4*>(&ldsA[arow][acol]);
      d[0] = pk.u[0];
      d[1] = pk.u[1];
    }
    // ---- stage B tile: pre-transposed bf16, contiguous async copies to LDS
    {
      const __bf16* gb = Bt + (size_t)(n0 + bn) * K + kt + bks;
      copy16_g2l(gb,     &ldsB[bn][bks]);
      copy16_g2l(gb + 8, &ldsB[bn][bks + 8]);
    }
    if (kt + TK < K) {  // global_prefetch_b8 for next K tile
      __builtin_prefetch(A + (size_t)(m0 + arow) * K + kt + TK + acol, 0, 1);
      __builtin_prefetch(Bt + (size_t)(n0 + bn) * K + kt + TK + bks, 0, 1);
    }
    wait_async();
    __syncthreads();

    v16bf af[2], bf[4];
    af[0] = load_fragA(&ldsA[wm * 32][0],      TK + PAD, lane);
    af[1] = load_fragA(&ldsA[wm * 32 + 16][0], TK + PAD, lane);
#pragma unroll
    for (int j = 0; j < 4; ++j)
      bf[j] = load_fragB(&ldsB[wn * 64 + j * 16][0], TK + PAD, lane);

#pragma unroll
    for (int i = 0; i < 2; ++i)
#pragma unroll
      for (int j = 0; j < 4; ++j)
        acc[i][j] = __builtin_amdgcn_wmma_f32_16x16x32_bf16(
            false, af[i], false, bf[j], (short)0, acc[i][j], false, false);
    __syncthreads();
  }

  // ---- epilogue (C layout: lanes 0-15 col n rows r; lanes 16-31 rows r+8)
  const int rbase = (lane >> 4) << 3;
  const int cidx  = lane & 15;
  if (mode == 0) {
    float* C = reinterpret_cast<float*>(Cout);
#pragma unroll
    for (int i = 0; i < 2; ++i)
#pragma unroll
      for (int j = 0; j < 4; ++j) {
        const int n = n0 + wn * 64 + j * 16 + cidx;
#pragma unroll
        for (int r = 0; r < 8; ++r) {
          const int m = m0 + wm * 32 + i * 16 + rbase + r;
          C[(size_t)m * N + n] = acc[i][j][r];
        }
      }
  } else {
    __bf16* C = reinterpret_cast<__bf16*>(Cout);
#pragma unroll
    for (int i = 0; i < 2; ++i)
#pragma unroll
      for (int j = 0; j < 4; ++j) {
        const int n = n0 + wn * 64 + j * 16 + cidx;
        const int h = n >> 6, d = n & 63;
#pragma unroll
        for (int r = 0; r < 8; ++r) {
          const int m = m0 + wm * 32 + i * 16 + rbase + r;
          const int b = m >> 11;  // / 2048
          const int s = m & 2047;
          const float val = acc[i][j][r] * scale;
          const size_t idx =
              (mode == 3) ? (((size_t)(b * cfg::Hh + h) * 64 + d) * cfg::Sq + s)
                          : (((size_t)(b * cfg::Hh + h) * cfg::Sq + s) * 64 + d);
          C[idx] = (__bf16)val;
        }
      }
  }
}

// ---------------------------------------------------------------------------
// Flash attention: grid (S/128, B*H), 256 threads = 8 waves.
// Wave w owns queries [q0, q0+16). 64-key blocks, double-buffered async K/V^T.
// ---------------------------------------------------------------------------
__global__ __launch_bounds__(256) void attn_wmma(
    const __bf16* __restrict__ q, const __bf16* __restrict__ k,
    const __bf16* __restrict__ vt, float* __restrict__ x) {
  constexpr int Sq = 2048, D = 64;
  __shared__ __align__(16) __bf16 ldsK[2][64][72];     // [buf][key][d]
  __shared__ __align__(16) __bf16 ldsV[2][64][72];     // [buf][dv][key]
  __shared__ __align__(16) __bf16 ldsP[8][16][72];     // per-wave P strip

  const int tid  = threadIdx.x;
  const int lane = tid & 31;
  const int wave = tid >> 5;
  const int bh   = blockIdx.y;  // b*H + h
  const int q0   = blockIdx.x * 128 + wave * 16;

  // Q fragments (A layout) straight from global [B,H,S,64] bf16.
  v16bf qf[2];
  {
    const int row = lane & 15;
    const int kb  = (lane >> 4) << 3;
    const __bf16* base = q + ((size_t)bh * Sq + q0 + row) * D;
#pragma unroll
    for (int j = 0; j < 2; ++j) {
      FragBF f;
      f.u[0] = *reinterpret_cast<const uint4*>(base + j * 32 + kb);
      f.u[1] = *reinterpret_cast<const uint4*>(base + j * 32 + kb + 16);
      qf[j] = f.v;
    }
  }

  v8f o[4];
#pragma unroll
  for (int t = 0; t < 4; ++t) o[t] = v8f_zero();
  float mrow[8], lrow[8];
#pragma unroll
  for (int r = 0; r < 8; ++r) { mrow[r] = -3.0e38f; lrow[r] = 0.0f; }

  const int krow  = tid >> 2;        // 0..63
  const int kseg  = (tid & 3) << 4;  // 0,16,32,48
  const int rbase = (lane >> 4) << 3;
  const int cidx  = lane & 15;

  auto issue_tile = [&](int blk, int buf) {
    const __bf16* ks = k + ((size_t)bh * Sq + blk * 64 + krow) * D + kseg;
    copy16_g2l(ks,     &ldsK[buf][krow][kseg]);
    copy16_g2l(ks + 8, &ldsK[buf][krow][kseg + 8]);
    const __bf16* vs = vt + ((size_t)bh * D + krow) * Sq + blk * 64 + kseg;
    copy16_g2l(vs,     &ldsV[buf][krow][kseg]);
    copy16_g2l(vs + 8, &ldsV[buf][krow][kseg + 8]);
  };

  issue_tile(0, 0);
  constexpr int NBLK = Sq / 64;
  for (int blk = 0; blk < NBLK; ++blk) {
    const int buf = blk & 1;
    wait_async();        // this wave's pending copies (block blk) landed
    __syncthreads();     // everyone's copies landed; prev buffer free
    if (blk + 1 < NBLK) issue_tile(blk + 1, buf ^ 1);  // overlap with compute

    // scores S = q @ k^T (scale folded into q): 4 key tiles x 2 k-steps
    v8f sc[4];
#pragma unroll
    for (int t = 0; t < 4; ++t) {
      sc[t] = v8f_zero();
      sc[t] = __builtin_amdgcn_wmma_f32_16x16x32_bf16(
          false, qf[0], false, load_fragB(&ldsK[buf][t * 16][0], 72, lane),
          (short)0, sc[t], false, false);
      sc[t] = __builtin_amdgcn_wmma_f32_16x16x32_bf16(
          false, qf[1], false, load_fragB(&ldsK[buf][t * 16][32], 72, lane),
          (short)0, sc[t], false, false);
    }

    // online softmax; rows live per (acc idx r, lane half), reduce over 16 lanes
#pragma unroll
    for (int r = 0; r < 8; ++r) {
      float mx = fmaxf(fmaxf(sc[0][r], sc[1][r]), fmaxf(sc[2][r], sc[3][r]));
#pragma unroll
      for (int off = 8; off >= 1; off >>= 1) mx = fmaxf(mx, __shfl_xor(mx, off, 32));
      const float mn    = fmaxf(mrow[r], mx);
      const float alpha = __expf(mrow[r] - mn);
      mrow[r] = mn;
      float ssum = 0.0f;
#pragma unroll
      for (int t = 0; t < 4; ++t) {
        const float p = __expf(sc[t][r] - mn);
        sc[t][r] = p;
        ssum += p;
      }
#pragma unroll
      for (int off = 8; off >= 1; off >>= 1) ssum += __shfl_xor(ssum, off, 32);
      lrow[r] = lrow[r] * alpha + ssum;
#pragma unroll
      for (int t = 0; t < 4; ++t) o[t][r] *= alpha;
    }

    // P (C layout) -> per-wave LDS strip -> re-fragment as A matrix
#pragma unroll
    for (int t = 0; t < 4; ++t)
#pragma unroll
      for (int r = 0; r < 8; ++r)
        ldsP[wave][rbase + r][t * 16 + cidx] = (__bf16)sc[t][r];

    const v16bf pf0 = load_fragA(&ldsP[wave][0][0],  72, lane);  // keys 0..31
    const v16bf pf1 = load_fragA(&ldsP[wave][0][32], 72, lane);  // keys 32..63

    // O += P @ V : 4 dv tiles x 2 key-steps
#pragma unroll
    for (int t = 0; t < 4; ++t) {
      o[t] = __builtin_amdgcn_wmma_f32_16x16x32_bf16(
          false, pf0, false, load_fragB(&ldsV[buf][t * 16][0], 72, lane),
          (short)0, o[t], false, false);
      o[t] = __builtin_amdgcn_wmma_f32_16x16x32_bf16(
          false, pf1, false, load_fragB(&ldsV[buf][t * 16][32], 72, lane),
          (short)0, o[t], false, false);
    }
  }

  // normalize and write x fp32 in [B,S,H*64] (merged-head layout for w_o GEMM)
  const int b = bh >> 4, h = bh & 15;
#pragma unroll
  for (int t = 0; t < 4; ++t)
#pragma unroll
    for (int r = 0; r < 8; ++r) {
      const int s = q0 + rbase + r;
      const int d = t * 16 + cidx;
      x[(size_t)(b * Sq + s) * 1024 + h * 64 + d] = o[t][r] / lrow[r];
    }
}

// ---------------------------------------------------------------------------
extern "C" void kernel_launch(void* const* d_in, const int* in_sizes, int n_in,
                              void* d_out, int out_size, void* d_ws, size_t ws_size,
                              hipStream_t stream) {
  const float* Q  = (const float*)d_in[0];
  const float* K  = (const float*)d_in[1];
  const float* V  = (const float*)d_in[2];
  const float* wq = (const float*)d_in[3];
  const float* wk = (const float*)d_in[4];
  const float* wv = (const float*)d_in[5];
  const float* wo = (const float*)d_in[6];

  constexpr int    Mrows = 4 * 2048;            // B*S = 8192
  constexpr int    Dm    = 1024;
  constexpr size_t elems = (size_t)Mrows * Dm;  // 8,388,608
  constexpr size_t wsz   = (size_t)Dm * Dm;     // 1,048,576

  char*   ws  = (char*)d_ws;
  __bf16* qb  = (__bf16*)(ws);                  // 16 MB
  __bf16* kb  = (__bf16*)(ws + 2 * elems);      // 16 MB
  __bf16* vtb = (__bf16*)(ws + 4 * elems);      // 16 MB
  float*  x   = (float*)(ws + 6 * elems);       // 32 MB fp32
  __bf16* wqT = (__bf16*)(ws + 10 * elems);               // 2 MB each
  __bf16* wkT = (__bf16*)(ws + 10 * elems + 2 * wsz);
  __bf16* wvT = (__bf16*)(ws + 10 * elems + 4 * wsz);
  __bf16* woT = (__bf16*)(ws + 10 * elems + 6 * wsz);

  const dim3 blk(256);
  // one-time weight transpose+convert (fp32 [K][N] -> bf16 [N][K])
  const dim3 tgrid(Dm / 32, Dm / 32);
  hipLaunchKernelGGL(transpose_convert, tgrid, blk, 0, stream, wq, wqT, Dm, Dm);
  hipLaunchKernelGGL(transpose_convert, tgrid, blk, 0, stream, wk, wkT, Dm, Dm);
  hipLaunchKernelGGL(transpose_convert, tgrid, blk, 0, stream, wv, wvT, Dm, Dm);
  hipLaunchKernelGGL(transpose_convert, tgrid, blk, 0, stream, wo, woT, Dm, Dm);

  const dim3 ggrid(Dm / 128, Mrows / 128);      // 8 x 64
  hipLaunchKernelGGL(gemm_bf16_wmma, ggrid, blk, 0, stream, Q, wqT, (void*)qb,
                     Mrows, Dm, Dm, 1, 0.125f);  // 1/sqrt(64) folded into q
  hipLaunchKernelGGL(gemm_bf16_wmma, ggrid, blk, 0, stream, K, wkT, (void*)kb,
                     Mrows, Dm, Dm, 1, 1.0f);
  hipLaunchKernelGGL(gemm_bf16_wmma, ggrid, blk, 0, stream, V, wvT, (void*)vtb,
                     Mrows, Dm, Dm, 3, 1.0f);

  const dim3 agrid(2048 / 128, 4 * 16);         // 16 x 64
  hipLaunchKernelGGL(attn_wmma, agrid, blk, 0, stream, qb, kb, vtb, x);

  hipLaunchKernelGGL(gemm_bf16_wmma, ggrid, blk, 0, stream, x, woT, d_out,
                     Mrows, Dm, Dm, 0, 1.0f);

  (void)in_sizes; (void)n_in; (void)out_size; (void)ws_size;
}
